// DilatedAttention_10634339025511
// MI455X (gfx1250) — matrix operand order repaired and checked by reading, not compiled, not run on hardware
//
#include <hip/hip_runtime.h>
#include <hip/hip_bf16.h>

// ---------------------------------------------------------------------------
// Dilated attention for MI455X (gfx1250): gather -> bf16 WMMA projections + LN
// -> flash-style attention (v_wmma_f32_16x16x32_bf16) -> feature softmax.
// ---------------------------------------------------------------------------

typedef __attribute__((ext_vector_type(16))) __bf16 bf16x16;
typedef __attribute__((ext_vector_type(8)))  __bf16 bf16x8;
typedef __attribute__((ext_vector_type(8)))  float  f32x8;

union ABFrag { bf16x16 v; bf16x8 h[2]; __bf16 e[16]; };

__device__ __forceinline__ f32x8 wmma_bf16(bf16x16 a, bf16x16 b, f32x8 c) {
  return __builtin_amdgcn_wmma_f32_16x16x32_bf16(false, a, false, b,
                                                 (short)0, c, false, false);
}

// ---------------------------------------------------------------------------
// Kernel 0: f32 -> bf16 weight downcast (vectorized float4 -> 4x bf16)
// ---------------------------------------------------------------------------
__global__ __launch_bounds__(256)
void cvt_f32_bf16(const float* __restrict__ s, __bf16* __restrict__ d, int n4) {
  int i = blockIdx.x * 256 + threadIdx.x;
  if (i < n4) {
    float4 x = ((const float4*)s)[i];
    __bf16* o = d + (size_t)i * 4;
    o[0] = (__bf16)x.x; o[1] = (__bf16)x.y;
    o[2] = (__bf16)x.z; o[3] = (__bf16)x.w;
  }
}

// ---------------------------------------------------------------------------
// Kernel 1: gather (dilated stride-4 within 2048-segments) + X @ W^T (bf16
// WMMA, f32 accum) + LayerNorm (mode 0: q/k) or transposed store (mode 1: v).
// grid = B * (2048/16) = 512 blocks, 256 threads (8 waves).
// Each wave computes 16 rows x 96 cols.
// ---------------------------------------------------------------------------
__global__ __launch_bounds__(256)
void proj_kernel(const float* __restrict__ X, const __bf16* __restrict__ W,
                 __bf16* __restrict__ out, const float* __restrict__ gamma,
                 const float* __restrict__ beta, int mode)
{
  __shared__ __bf16 sA[16][800];    // gathered rows, bf16
  __shared__ float  sOut[16][776];  // f32 accumulators for LN / transpose

  const int tid  = threadIdx.x;
  const int b    = blockIdx.x >> 7;      // batch
  const int tile = blockIdx.x & 127;
  const int r0   = tile << 4;            // first sparse row of this tile

  // --- gather + convert: 16 rows x 768 f32 -> bf16 in LDS -----------------
  #pragma unroll
  for (int j = 0; j < 12; ++j) {
    int i   = tid + 256 * j;             // float4 index 0..3071
    int row = i / 192;
    int c4  = (i % 192) * 4;
    int sr  = r0 + row;                  // sparse row in [0,2048)
    int grow = ((sr >> 9) << 11) + ((sr & 511) << 2);   // seg*2048 + in*4
    float4 x = *(const float4*)(X + ((size_t)b * 8192 + grow) * 768 + c4);
    sA[row][c4 + 0] = (__bf16)x.x;
    sA[row][c4 + 1] = (__bf16)x.y;
    sA[row][c4 + 2] = (__bf16)x.z;
    sA[row][c4 + 3] = (__bf16)x.w;
  }
  __syncthreads();

  const int wave  = tid >> 5;
  const int lane  = tid & 31;
  const int lhalf = lane >> 4;
  const int l15   = lane & 15;
  const int nbase = wave * 96;

  f32x8 acc[6];
  #pragma unroll
  for (int t = 0; t < 6; ++t)
    #pragma unroll
    for (int v = 0; v < 8; ++v) acc[t][v] = 0.0f;

  for (int kt = 0; kt < 24; ++kt) {
    ABFrag a;
    int ka = kt * 32 + lhalf * 16;
    a.h[0] = *(const bf16x8*)&sA[l15][ka];
    a.h[1] = *(const bf16x8*)&sA[l15][ka + 8];
    int kb = kt * 32 + lhalf * 16;
    #pragma unroll
    for (int t = 0; t < 6; ++t) {
      int n = nbase + t * 16 + l15;
      bf16x16 bf = *(const bf16x16*)(W + (size_t)n * 768 + kb);  // W[n][k]
      acc[t] = wmma_bf16(a.v, bf, acc[t]);
    }
  }

  #pragma unroll
  for (int t = 0; t < 6; ++t)
    #pragma unroll
    for (int v = 0; v < 8; ++v)
      sOut[v + lhalf * 8][nbase + t * 16 + l15] = acc[t][v];
  __syncthreads();

  if (mode == 0) {
    // ---- LayerNorm per row, write row-major bf16 -------------------------
    #pragma unroll
    for (int rr = 0; rr < 2; ++rr) {
      int row = wave * 2 + rr;
      float s = 0.f, s2 = 0.f;
      #pragma unroll
      for (int j = 0; j < 24; ++j) {
        float x = sOut[row][lane + 32 * j];
        s += x; s2 += x * x;
      }
      #pragma unroll
      for (int m = 16; m >= 1; m >>= 1) {
        s  += __shfl_xor(s,  m, 32);
        s2 += __shfl_xor(s2, m, 32);
      }
      float mu   = s * (1.0f / 768.0f);
      float var  = s2 * (1.0f / 768.0f) - mu * mu;
      float rstd = rsqrtf(var + 1e-5f);
      size_t obase = ((size_t)(b * 2048 + r0 + row)) * 768;
      #pragma unroll
      for (int j = 0; j < 24; ++j) {
        int col = lane + 32 * j;
        float x = (sOut[row][col] - mu) * rstd * gamma[col] + beta[col];
        out[obase + col] = (__bf16)x;
      }
    }
  } else {
    // ---- transposed store: vT[b][d][key], 32B contiguous per column ------
    for (int cc = tid; cc < 768; cc += 256) {
      ABFrag t16;
      #pragma unroll
      for (int r = 0; r < 16; ++r) t16.e[r] = (__bf16)sOut[r][cc];
      *(bf16x16*)(out + ((size_t)(b * 768 + cc)) * 2048 + r0) = t16.v;
    }
  }
}

// ---------------------------------------------------------------------------
// Kernel 2: flash attention + final feature-dim softmax.
// grid = B * 128 blocks of 256 threads. Each block owns 16 query rows.
// Per key-block of 128 keys: each wave computes a 16x16 score tile (K=768),
// online softmax across waves via LDS, P exchanged as bf16 through LDS,
// O (16x768, distributed 96 cols/wave) updated via WMMA against vT.
// ---------------------------------------------------------------------------
__global__ __launch_bounds__(256)
void attn_kernel(const __bf16* __restrict__ qn, const __bf16* __restrict__ kn,
                 const __bf16* __restrict__ vT, float* __restrict__ out)
{
  __shared__ __bf16 sQ[16][800];
  __shared__ __bf16 sP[16][136];
  __shared__ float  sRmax[16][8];
  __shared__ float  sRsum[16][8];
  __shared__ float  sM[16];
  __shared__ float  sL[16];
  __shared__ float  sFac[16];
  __shared__ float  sO[16][776];

  const int tid  = threadIdx.x;
  const int b    = blockIdx.x >> 7;
  const int tile = blockIdx.x & 127;
  const int r0   = tile << 4;

  #pragma unroll
  for (int j = 0; j < 6; ++j) {
    int i   = tid + 256 * j;            // 8-elem chunk index 0..1535
    int row = i / 96;
    int c8  = (i % 96) * 8;
    *(bf16x8*)&sQ[row][c8] =
        *(const bf16x8*)(qn + ((size_t)(b * 2048 + r0 + row)) * 768 + c8);
  }
  if (tid < 16) { sM[tid] = -3.0e38f; sL[tid] = 0.0f; }
  __syncthreads();

  const int wave  = tid >> 5;
  const int lane  = tid & 31;
  const int lhalf = lane >> 4;
  const int l15   = lane & 15;

  f32x8 o[6];
  #pragma unroll
  for (int t = 0; t < 6; ++t)
    #pragma unroll
    for (int v = 0; v < 8; ++v) o[t][v] = 0.0f;

  const float scale = 0.03608439182435161f;  // 1/sqrt(768)

  for (int kb = 0; kb < 16; ++kb) {
    // ---- S = q . k^T for this wave's 16 keys -----------------------------
    f32x8 s;
    #pragma unroll
    for (int v = 0; v < 8; ++v) s[v] = 0.0f;
    const int key = kb * 128 + wave * 16 + l15;
    for (int kt = 0; kt < 24; ++kt) {
      ABFrag a;
      int ka = kt * 32 + lhalf * 16;
      a.h[0] = *(const bf16x8*)&sQ[l15][ka];
      a.h[1] = *(const bf16x8*)&sQ[l15][ka + 8];
      int kd = kt * 32 + lhalf * 16;
      bf16x16 bf = *(const bf16x16*)(kn + ((size_t)(b * 2048 + key)) * 768 + kd);
      s = wmma_bf16(a.v, bf, s);
    }
    #pragma unroll
    for (int v = 0; v < 8; ++v) s[v] *= scale;

    // ---- per-row max within fragment (16-lane halves) --------------------
    #pragma unroll
    for (int v = 0; v < 8; ++v) {
      float x = s[v];
      x = fmaxf(x, __shfl_xor(x, 1, 32));
      x = fmaxf(x, __shfl_xor(x, 2, 32));
      x = fmaxf(x, __shfl_xor(x, 4, 32));
      x = fmaxf(x, __shfl_xor(x, 8, 32));
      if (l15 == 0) sRmax[v + lhalf * 8][wave] = x;
    }
    __syncthreads();
    if (tid < 16) {
      float bm = sRmax[tid][0];
      #pragma unroll
      for (int w = 1; w < 8; ++w) bm = fmaxf(bm, sRmax[tid][w]);
      float mold = sM[tid];
      float mnew = fmaxf(mold, bm);
      float fac  = __expf(mold - mnew);
      sM[tid] = mnew; sFac[tid] = fac; sL[tid] *= fac;
    }
    __syncthreads();

    float mrow[8], frow[8];
    #pragma unroll
    for (int v = 0; v < 8; ++v) {
      mrow[v] = sM[lhalf * 8 + v];
      frow[v] = sFac[lhalf * 8 + v];
    }
    #pragma unroll
    for (int t = 0; t < 6; ++t)
      #pragma unroll
      for (int v = 0; v < 8; ++v) o[t][v] *= frow[v];

    // ---- P = exp(S - m), exchange via LDS, row sums ----------------------
    #pragma unroll
    for (int v = 0; v < 8; ++v) {
      float p = __expf(s[v] - mrow[v]);
      sP[lhalf * 8 + v][wave * 16 + l15] = (__bf16)p;
      float x = p;
      x += __shfl_xor(x, 1, 32);
      x += __shfl_xor(x, 2, 32);
      x += __shfl_xor(x, 4, 32);
      x += __shfl_xor(x, 8, 32);
      if (l15 == 0) sRsum[v + lhalf * 8][wave] = x;
    }
    __syncthreads();
    if (tid < 16) {
      float t = 0.f;
      #pragma unroll
      for (int w = 0; w < 8; ++w) t += sRsum[tid][w];
      sL[tid] += t;
    }

    // ---- O += P . V (vT gives contiguous B-fragment loads) ---------------
    #pragma unroll
    for (int kt2 = 0; kt2 < 4; ++kt2) {
      ABFrag a;
      int ka = kt2 * 32 + lhalf * 16;
      a.h[0] = *(const bf16x8*)&sP[l15][ka];
      a.h[1] = *(const bf16x8*)&sP[l15][ka + 8];
      int krow = kb * 128 + kt2 * 32 + lhalf * 16;
      #pragma unroll
      for (int t = 0; t < 6; ++t) {
        int n = wave * 96 + t * 16 + l15;
        bf16x16 bf =
            *(const bf16x16*)(vT + ((size_t)(b * 768 + n)) * 2048 + krow);
        o[t] = wmma_bf16(a.v, bf, o[t]);
      }
    }
    __syncthreads();
  }

  // ---- O /= l, then feature-dim softmax, write f32 out -------------------
  float linv[8];
  #pragma unroll
  for (int v = 0; v < 8; ++v) linv[v] = 1.0f / sL[lhalf * 8 + v];
  #pragma unroll
  for (int t = 0; t < 6; ++t)
    #pragma unroll
    for (int v = 0; v < 8; ++v)
      sO[v + lhalf * 8][wave * 96 + t * 16 + l15] = o[t][v] * linv[v];
  __syncthreads();

  #pragma unroll
  for (int rr = 0; rr < 2; ++rr) {
    int row = wave * 2 + rr;
    float mx = -3.0e38f;
    #pragma unroll
    for (int j = 0; j < 24; ++j) mx = fmaxf(mx, sO[row][lane + 32 * j]);
    mx = fmaxf(mx, __shfl_xor(mx, 1, 32));
    mx = fmaxf(mx, __shfl_xor(mx, 2, 32));
    mx = fmaxf(mx, __shfl_xor(mx, 4, 32));
    mx = fmaxf(mx, __shfl_xor(mx, 8, 32));
    mx = fmaxf(mx, __shfl_xor(mx, 16, 32));
    float se = 0.f;
    #pragma unroll
    for (int j = 0; j < 24; ++j) se += __expf(sO[row][lane + 32 * j] - mx);
    se += __shfl_xor(se, 1, 32);
    se += __shfl_xor(se, 2, 32);
    se += __shfl_xor(se, 4, 32);
    se += __shfl_xor(se, 8, 32);
    se += __shfl_xor(se, 16, 32);
    float inv = 1.0f / se;
    size_t obase = ((size_t)(b * 2048 + r0 + row)) * 768;
    #pragma unroll
    for (int j = 0; j < 24; ++j) {
      int col = lane + 32 * j;
      out[obase + col] = __expf(sO[row][col] - mx) * inv;
    }
  }
}

// ---------------------------------------------------------------------------
// Host launcher. ws layout (bf16 elements):
//   wq/wk/wv: 3 * 589824,  qn/kn: 2 * 4*2048*768,  vT: 4*768*2048
// total ~41.3 MB.
// ---------------------------------------------------------------------------
extern "C" void kernel_launch(void* const* d_in, const int* in_sizes, int n_in,
                              void* d_out, int out_size, void* d_ws,
                              size_t ws_size, hipStream_t stream) {
  (void)in_sizes; (void)n_in; (void)out_size; (void)ws_size;

  const float* Q     = (const float*)d_in[0];
  const float* K     = (const float*)d_in[1];
  const float* V     = (const float*)d_in[2];
  const float* Wq    = (const float*)d_in[3];
  const float* Wk    = (const float*)d_in[4];
  const float* Wv    = (const float*)d_in[5];
  const float* gamma = (const float*)d_in[6];
  const float* beta  = (const float*)d_in[7];
  float* outp        = (float*)d_out;

  __bf16* wq  = (__bf16*)d_ws;
  __bf16* wk  = wq + 589824;           // 768*768
  __bf16* wv  = wk + 589824;
  __bf16* qn  = wv + 589824;
  __bf16* knb = qn + 6291456;          // 4*2048*768
  __bf16* vT  = knb + 6291456;

  cvt_f32_bf16<<<576, 256, 0, stream>>>(Wq, wq, 147456);
  cvt_f32_bf16<<<576, 256, 0, stream>>>(Wk, wk, 147456);
  cvt_f32_bf16<<<576, 256, 0, stream>>>(Wv, wv, 147456);

  proj_kernel<<<512, 256, 0, stream>>>(Q, wq, qn,  gamma, beta, 0);
  proj_kernel<<<512, 256, 0, stream>>>(K, wk, knb, gamma, beta, 0);
  proj_kernel<<<512, 256, 0, stream>>>(V, wv, vT,  gamma, beta, 1);

  attn_kernel<<<512, 256, 0, stream>>>(qn, knb, vT, outp);
}
